// FixedRevRNN_41884521070982
// MI455X (gfx1250) — matrix operand order repaired and checked by reading, not compiled, not run on hardware
//
#include <hip/hip_runtime.h>

// ---------------------------------------------------------------------------
// FixedRevRNN on MI455X (gfx1250, wave32, WMMA)
//
// 264-step sequential scan => latency bound (HBM traffic ~70MB vs 23.3TB/s).
// Persistent kernel: 8 workgroups x 16 batch rows (WMMA M=16). State and
// activations in LDS; weights pre-converted once to bf16 transposed [N][K]
// (1MB, L2-resident) and streamed every step. Anti-LICM pointer laundering
// (asm "+s") prevents cross-step hoisting of weight fragments (v1 spilled);
// explicit addrspace(1) casts at the load sites keep them on the
// global_load_b128 path (v2 regressed to flat_load, ticking DScnt).
// The two leading GEMMs of each _calc are fused: their accumulators share
// the same C/D lane mapping, so u = relu(xA * sB) is formed in registers and
// written once as bf16 (removes 64KB/step of f32 LDS round-trips).
// GEMMs: v_wmma_f32_16x16x32_bf16. Per-batch 16x16 h@Q:
// v_wmma_f32_16x16x4_f32. QR(16x16): modified Gram-Schmidt, lane-per-row.
// ---------------------------------------------------------------------------

typedef __attribute__((ext_vector_type(16))) __bf16       v16bf;
typedef __attribute__((ext_vector_type(8)))  float        v8f;
typedef __attribute__((ext_vector_type(2)))  float        v2f;
typedef __attribute__((ext_vector_type(4)))  unsigned int u32x4;

#define GLOBAL __attribute__((address_space(1)))
typedef const GLOBAL u32x4* gvec_p;

#define FEAT   256
#define TLEN   256
#define NSTEP  264   // TLEN + DELAY
#define DELAY  8
#define BPW    16    // batch rows per workgroup (WMMA M)

#define WMMA_BF16(A, B, C) \
  __builtin_amdgcn_wmma_f32_16x16x32_bf16(false, (A), false, (B), (short)0, (C), false, false)

#if defined(__has_builtin)
#if __has_builtin(__builtin_amdgcn_wmma_f32_16x16x4_f32)
#define HAVE_WMMA_F32X4 1
#endif
#endif

// Launder a uniform pointer through an empty asm: the compiler must assume it
// changes every call, so loads through it cannot be hoisted/CSE'd across the
// sequential step loop (prevents the v1 register-spill explosion).
__device__ __forceinline__ const __bf16* opaque(const __bf16* p) {
  asm volatile("" : "+s"(p));
  return p;
}

// sum across 16-lane half-wave (xor masks 1,2,4,8 keep halves independent)
__device__ __forceinline__ float hsum16(float v) {
  v += __shfl_xor(v, 1);
  v += __shfl_xor(v, 2);
  v += __shfl_xor(v, 4);
  v += __shfl_xor(v, 8);
  return v;
}

// A operand fragment: 16x32 bf16, lda = 256 (LDS activation tile).
// Lane<16 (row=lr): K = k0+0..15
// Lane>=16(row=lr): K = k0+16..31
__device__ __forceinline__ v16bf load_a_frag(const __bf16* a, int lr, int half, int k0) {
  union { v16bf v; u32x4 u[2]; } f;
  const u32x4* p = (const u32x4*)(a + lr * FEAT + k0 + half * 16);
  f.u[0] = p[0];   // 8 bf16
  f.u[1] = p[1];   // +8 elements
  return f.v;
}

// B operand fragment: 32x16 bf16 from transposed weight [N][ldw] in GLOBAL
// memory (forced global_load_b128: pure LOADcnt, no flat/DScnt involvement).
// Lane<16 (col=n): K = k..k+15 ; lane>=16: K = k+16..k+31 (contiguous in wT)
__device__ __forceinline__ v16bf load_b_frag(const __bf16* wt, int ldw, int col,
                                             int half, int k) {
  union { v16bf v; u32x4 u[2]; } f;
  gvec_p p = (gvec_p)(unsigned long long)(wt + (size_t)col * ldw + k + half * 16);
  f.u[0] = p[0];
  f.u[1] = p[1];
  return f.v;
}

// C = A(16x256, bf16 LDS) @ W(256x256 transposed bf16) -> f32 LDS.
// Each of 8 waves computes 2 adjacent 16x16 N-tiles (columns 32w..32w+31).
__device__ __forceinline__ void gemm_lds(const __bf16* __restrict__ a,
                                         const __bf16* wt_in,
                                         float* __restrict__ c,
                                         int wv, int lr, int half) {
  const __bf16* wt = opaque(wt_in);
  const int n0 = wv * 32;
  v8f acc0 = {}, acc1 = {};
#pragma unroll
  for (int k0 = 0; k0 < FEAT; k0 += 32) {
    v16bf fa  = load_a_frag(a, lr, half, k0);
    v16bf fb0 = load_b_frag(wt, FEAT, n0 + lr,      half, k0);
    v16bf fb1 = load_b_frag(wt, FEAT, n0 + 16 + lr, half, k0);
    acc0 = WMMA_BF16(fa, fb0, acc0);
    acc1 = WMMA_BF16(fa, fb1, acc1);
  }
#pragma unroll
  for (int i = 0; i < 8; ++i) {
    const int row = i + half * 8;  // D layout: VGPR i -> M=i (lanes 0-15) / M=8+i
    c[row * FEAT + n0 + lr]      = acc0[i];
    c[row * FEAT + n0 + 16 + lr] = acc1[i];
  }
}

// Fused: dst = bf16(relu((a1 @ W1) * (a2 @ W2))), all elementwise work done in
// the WMMA accumulators (identical C/D lane mapping for both products).
// dst may alias a1: an internal barrier separates the read and write phases.
// Must be called uniformly by all threads of the block.
__device__ __forceinline__ void gemm2_mul_relu(const __bf16* __restrict__ a1,
                                               const __bf16* w1_in,
                                               const __bf16* __restrict__ a2,
                                               const __bf16* w2_in,
                                               __bf16* dst,
                                               int wv, int lr, int half) {
  const __bf16* w1 = opaque(w1_in);
  const __bf16* w2 = opaque(w2_in);
  const int n0 = wv * 32;
  v8f p0 = {}, p1 = {}, q0 = {}, q1 = {};
#pragma unroll
  for (int k0 = 0; k0 < FEAT; k0 += 32) {
    v16bf fa1 = load_a_frag(a1, lr, half, k0);
    v16bf fa2 = load_a_frag(a2, lr, half, k0);
    v16bf fb0 = load_b_frag(w1, FEAT, n0 + lr,      half, k0);
    v16bf fb1 = load_b_frag(w1, FEAT, n0 + 16 + lr, half, k0);
    v16bf fb2 = load_b_frag(w2, FEAT, n0 + lr,      half, k0);
    v16bf fb3 = load_b_frag(w2, FEAT, n0 + 16 + lr, half, k0);
    p0 = WMMA_BF16(fa1, fb0, p0);
    p1 = WMMA_BF16(fa1, fb1, p1);
    q0 = WMMA_BF16(fa2, fb2, q0);
    q1 = WMMA_BF16(fa2, fb3, q1);
  }
  __syncthreads();   // every wave finished reading a1 before dst(=a1) is written
#pragma unroll
  for (int i = 0; i < 8; ++i) {
    const int row = i + half * 8;
    dst[row * FEAT + n0 + lr]      = (__bf16)fmaxf(p0[i] * q0[i], 0.0f);
    dst[row * FEAT + n0 + 16 + lr] = (__bf16)fmaxf(p1[i] * q1[i], 0.0f);
  }
}

// QR (Q factor) of each 16x16 matrix via modified Gram-Schmidt.
// Wave w handles batch matrices 2w (lanes 0-15) and 2w+1 (lanes 16-31);
// each lane owns one row. Fully unrolled -> stays in VGPRs.
__device__ __forceinline__ void qr16(const float* __restrict__ o,
                                     float* __restrict__ qout,
                                     int wv, int lr, int half) {
  const int b = wv * 2 + half;
  const float* src = o + b * FEAT + lr * 16;   // row lr of matrix b
  float q[16];
#pragma unroll
  for (int j = 0; j < 16; ++j) {
    float v = src[j];                          // column j, this row
#pragma unroll
    for (int i = 0; i < 16; ++i) {
      if (i < j) {
        float r = hsum16(q[i] * v);
        v -= r * q[i];
      }
    }
    float n2  = hsum16(v * v);
    float inv = rsqrtf(fmaxf(n2, 1e-30f));
    q[j] = v * inv;
  }
  float* dst = qout + b * FEAT + lr * 16;
#pragma unroll
  for (int c = 0; c < 16; ++c) dst[c] = q[c];
}

// Per-batch P = H(16x16) @ Q(16x16) in exact f32. Updates h in place and
// emits raw-bf16 (for the output projection) + relu-bf16 (next GEMM input).
// Wave-internal only: each wave owns batches 2w, 2w+1 -> no barrier needed
// (all lanes' loads are issued before any lane's store: single wave PC).
__device__ __forceinline__ void small_mm(float* __restrict__ h,
                                         const float* __restrict__ qm,
                                         __bf16* __restrict__ ybf,
                                         __bf16* __restrict__ xbf,
                                         int wv, int lr, int half) {
#pragma unroll
  for (int j = 0; j < 2; ++j) {
    const int b = wv * 2 + j;
    const float* H = h + b * FEAT;
    const float* Q = qm + b * FEAT;
#if defined(HAVE_WMMA_F32X4)
    v8f acc = {};
#pragma unroll
    for (int k0 = 0; k0 < 16; k0 += 4) {
      // A 16x4 f32: lane<16 K={k0,k0+1}; lane>=16 K={k0+2,k0+3}
      v2f fa, fb;
      fa.x = H[lr * 16 + k0 + 2 * half];
      fa.y = H[lr * 16 + k0 + 2 * half + 1];
      // B 4x16 f32: lane<16 K={k0,k0+1}, N=lane; lane>=16 K={k0+2,k0+3}
      fb.x = Q[(k0 + 2 * half) * 16 + lr];
      fb.y = Q[(k0 + 2 * half + 1) * 16 + lr];
      acc = __builtin_amdgcn_wmma_f32_16x16x4_f32(false, fa, false, fb,
                                                  (short)0, acc, false, false);
    }
#pragma unroll
    for (int i = 0; i < 8; ++i) {
      const int row = i + half * 8;
      const float v = acc[i];
      h[b * FEAT + row * 16 + lr]   = v;                       // new hidden state
      ybf[b * FEAT + row * 16 + lr] = (__bf16)v;               // raw for y
      xbf[b * FEAT + row * 16 + lr] = (__bf16)fmaxf(v, 0.0f);  // relu for next calc
    }
#else
    float res[8];
#pragma unroll
    for (int i = 0; i < 8; ++i) {
      const int row = i + half * 8;
      float acc = 0.0f;
#pragma unroll
      for (int k = 0; k < 16; ++k) acc += H[row * 16 + k] * Q[k * 16 + lr];
      res[i] = acc;
    }
#pragma unroll
    for (int i = 0; i < 8; ++i) {
      const int row = i + half * 8;
      h[b * FEAT + row * 16 + lr]   = res[i];
      ybf[b * FEAT + row * 16 + lr] = (__bf16)res[i];
      xbf[b * FEAT + row * 16 + lr] = (__bf16)fmaxf(res[i], 0.0f);
    }
#endif
  }
}

// out[b][t-8][:] = y0 @ Wout[0:256,:] + y1 @ Wout[256:512,:]
// Wout transposed bf16 [256 n][512 k]. Output stored nontemporal (write-once).
__device__ __forceinline__ void proj_out(const __bf16* __restrict__ y0,
                                         const __bf16* __restrict__ y1,
                                         const __bf16* wt_in,
                                         float* __restrict__ out,
                                         int b0, int trow,
                                         int wv, int lr, int half) {
  const __bf16* wt = opaque(wt_in);
  const int n0 = wv * 32;
  v8f acc0 = {}, acc1 = {};
#pragma unroll
  for (int k0 = 0; k0 < FEAT; k0 += 32) {
    v16bf fa  = load_a_frag(y0, lr, half, k0);
    v16bf fb0 = load_b_frag(wt, 512, n0 + lr,      half, k0);
    v16bf fb1 = load_b_frag(wt, 512, n0 + 16 + lr, half, k0);
    acc0 = WMMA_BF16(fa, fb0, acc0);
    acc1 = WMMA_BF16(fa, fb1, acc1);
  }
#pragma unroll
  for (int k0 = 0; k0 < FEAT; k0 += 32) {
    v16bf fa  = load_a_frag(y1, lr, half, k0);
    v16bf fb0 = load_b_frag(wt, 512, n0 + lr,      half, 256 + k0);
    v16bf fb1 = load_b_frag(wt, 512, n0 + 16 + lr, half, 256 + k0);
    acc0 = WMMA_BF16(fa, fb0, acc0);
    acc1 = WMMA_BF16(fa, fb1, acc1);
  }
#pragma unroll
  for (int i = 0; i < 8; ++i) {
    const int row = i + half * 8;
    const size_t base = (((size_t)(b0 + row)) * TLEN + trow) * FEAT;
    __builtin_nontemporal_store(acc0[i], out + base + n0 + lr);
    __builtin_nontemporal_store(acc1[i], out + base + n0 + 16 + lr);
  }
}

// ---------------------------------------------------------------------------
// Prepass: convert weights to bf16, transposed [N][K], into workspace.
// ws layout (bf16): 6 x 256x256 (l0a l0b l0c l1a l1b l1c), then 256x512 WoutT.
// ---------------------------------------------------------------------------
__global__ void prep_weights(const float* __restrict__ l0a, const float* __restrict__ l0b,
                             const float* __restrict__ l0c, const float* __restrict__ l1a,
                             const float* __restrict__ l1b, const float* __restrict__ l1c,
                             const float* __restrict__ outl, __bf16* __restrict__ wts) {
  const int idx = blockIdx.x * 256 + threadIdx.x;   // 0..524287
  if (idx < 6 * 65536) {
    const int w = idx >> 16, r = idx & 65535, n = r >> 8, k = r & 255;
    const float* src;
    switch (w) {
      case 0:  src = l0a; break;
      case 1:  src = l0b; break;
      case 2:  src = l0c; break;
      case 3:  src = l1a; break;
      case 4:  src = l1b; break;
      default: src = l1c; break;
    }
    wts[idx] = (__bf16)src[k * FEAT + n];
  } else {
    const int r = idx - 6 * 65536, n = r >> 9, k = r & 511;  // [n][k], ldk=512
    wts[idx] = (__bf16)outl[k * FEAT + n];
  }
}

// ---------------------------------------------------------------------------
// Persistent recurrence kernel: 8 blocks x 256 threads (8 wave32 / WGP).
// Dynamic LDS: h0,h1,t0,t1 (f32 16x256 each) + A_bf,s_bf,y0,y1 (bf16) = 96KB.
// ---------------------------------------------------------------------------
__global__ __launch_bounds__(256, 1)
void rnn_persistent(const int* __restrict__ toks, const float* __restrict__ emb,
                    const float* __restrict__ h0init, const __bf16* __restrict__ wts,
                    float* __restrict__ out) {
  extern __shared__ char smem_raw[];
  float*  h0   = (float*)smem_raw;          // [16][256]
  float*  h1   = h0 + BPW * FEAT;
  float*  t0   = h1 + BPW * FEAT;
  float*  t1   = t0 + BPW * FEAT;
  __bf16* A_bf = (__bf16*)(t1 + BPW * FEAT);  // GEMM A operand (relu'd state / u)
  __bf16* s_bf = A_bf + BPW * FEAT;           // embedding row (bf16)
  __bf16* y0   = s_bf + BPW * FEAT;           // raw o0 (for projection)
  __bf16* y1   = y0 + BPW * FEAT;             // raw o1

  const int tid  = threadIdx.x;
  const int wv   = tid >> 5;
  const int ln   = tid & 31;
  const int half = ln >> 4;
  const int lr   = ln & 15;
  const int b0   = blockIdx.x * BPW;

  const __bf16* w0a  = wts;
  const __bf16* w0b  = wts + 1 * 65536;
  const __bf16* w0c  = wts + 2 * 65536;
  const __bf16* w1a  = wts + 3 * 65536;
  const __bf16* w1b  = wts + 4 * 65536;
  const __bf16* w1c  = wts + 5 * 65536;
  const __bf16* wout = wts + 6 * 65536;   // [256][512] transposed

  // init: both hidden states = broadcast hidden_state0; A_bf = relu(h0)
  for (int i = tid; i < BPW * FEAT; i += 256) {
    const float v = h0init[i & (FEAT - 1)];
    h0[i] = v;
    h1[i] = v;
    A_bf[i] = (__bf16)fmaxf(v, 0.0f);
  }
  __syncthreads();

  const int srow = tid >> 4;            // 0..15 batch row for s loading
  const int sc0  = (tid & 15) * 16;     // 16 floats per thread

  for (int t = 0; t < NSTEP; ++t) {
    // ---- load s_t = embedding[token] for our 16 batch rows, cvt to bf16 ----
    {
      const GLOBAL int* tp = (const GLOBAL int*)(unsigned long long)
                             (toks + (b0 + srow) * TLEN + t);
      const int tok = (t < TLEN) ? *tp : 0;
      const GLOBAL float* er = (const GLOBAL float*)(unsigned long long)
                               (emb + (size_t)tok * FEAT + sc0);
#pragma unroll
      for (int i = 0; i < 16; ++i) s_bf[srow * FEAT + sc0 + i] = (__bf16)er[i];
      if (t + 1 < TLEN) {  // prefetch next step's row (global_prefetch_b8)
        const int ntok = toks[(b0 + srow) * TLEN + t + 1];
        __builtin_prefetch(emb + (size_t)ntok * FEAT + sc0, 0, 1);
      }
    }
    __syncthreads();

    // ================= _calc(h0, s, l0a,l0b,l0c) =================
    // A_bf = relu(h0) already; fused u = relu((x@l0a)*(s@l0b)) -> A_bf
    gemm2_mul_relu(A_bf, w0a, s_bf, w0b, A_bf, wv, lr, half);
    __syncthreads();
    gemm_lds(A_bf, w0c, t0, wv, lr, half);   // o = u @ l0c
    __syncthreads();
    qr16(t0, t1, wv, lr, half);              // Q0 -> t1
    __syncthreads();
    // o1 = h1 @ Q0 ; h1 <- o1 ; y1 = bf16(o1) ; A_bf = relu(o1)
    small_mm(h1, t1, y1, A_bf, wv, lr, half);
    __syncthreads();

    // ================= _calc(o1, s, l1a,l1b,l1c) =================
    gemm2_mul_relu(A_bf, w1a, s_bf, w1b, A_bf, wv, lr, half);
    __syncthreads();
    gemm_lds(A_bf, w1c, t0, wv, lr, half);
    __syncthreads();
    qr16(t0, t1, wv, lr, half);              // Q1 -> t1
    __syncthreads();
    // o0 = h0 @ Q1 ; h0 <- o0 ; y0 = bf16(o0) ; A_bf = relu(o0) (next step x)
    small_mm(h0, t1, y0, A_bf, wv, lr, half);
    __syncthreads();

    // ============ fused output projection (skip first DELAY steps) ============
    if (t >= DELAY) proj_out(y0, y1, wout, out, b0, t - DELAY, wv, lr, half);
    __syncthreads();
  }
}

// ---------------------------------------------------------------------------
extern "C" void kernel_launch(void* const* d_in, const int* in_sizes, int n_in,
                              void* d_out, int out_size, void* d_ws, size_t ws_size,
                              hipStream_t stream) {
  (void)in_sizes; (void)n_in; (void)out_size; (void)ws_size;

  const int*   toks = (const int*)d_in[0];
  const float* l0a  = (const float*)d_in[1];
  const float* l0b  = (const float*)d_in[2];
  const float* l0c  = (const float*)d_in[3];
  const float* l1a  = (const float*)d_in[4];
  const float* l1b  = (const float*)d_in[5];
  const float* l1c  = (const float*)d_in[6];
  const float* emb  = (const float*)d_in[7];
  const float* outl = (const float*)d_in[8];   // [1][512][256]
  const float* h0i  = (const float*)d_in[9];   // [1][16][16]
  float*       out  = (float*)d_out;
  __bf16*      wts  = (__bf16*)d_ws;           // needs 1MB (6*64K + 128K bf16)

  // 1) one-shot weight convert+transpose to bf16 (re-run every call: stateless)
  prep_weights<<<2048, 256, 0, stream>>>(l0a, l0b, l0c, l1a, l1b, l1c, outl, wts);

  // 2) persistent recurrence: 8 WGs x 16 batch rows, 96KB dynamic LDS each
  constexpr size_t kLds = (4 * sizeof(float) + 4 * sizeof(__bf16)) * BPW * FEAT;
  rnn_persistent<<<8, 256, kLds, stream>>>(toks, emb, h0i, wts, out);
}
